// DispersionLoss_40596030882405
// MI455X (gfx1250) — matrix operand order repaired and checked by reading, not compiled, not run on hardware
//
#include <hip/hip_runtime.h>

// MI455X / gfx1250, wave32. Pairwise-distance loss:
//   G = Z Z^T via v_wmma_f32_16x16x32_f16, epilogue exp(-sqrt(sq_i+sq_j-2g)),
//   upper-triangular 128x128 tiles (~34 GFLOP). Compute-bound (Z = 16MB, L2-resident).
// PRE path: f16 Z + row norms precomputed once into d_ws; GEMM stages use
// GLOBAL_LOAD_ASYNC_TO_LDS_B128 with double-buffered LDS (ASYNCcnt pipelining).

typedef _Float16 v16h __attribute__((ext_vector_type(16)));
typedef _Float16 v8h  __attribute__((ext_vector_type(8)));
typedef _Float16 v4h  __attribute__((ext_vector_type(4)));
typedef float    v8f  __attribute__((ext_vector_type(8)));

#define TILE     128
#define BK       64
#define LDSTR    72                    // padded LDS row stride in halves
#define BUFH     (2 * TILE * LDSTR)    // halves per double-buffer slot (A+B)
#define NTHREADS 256

__global__ void disp_zero_out(float* out) { out[0] = 0.0f; }

// One wave per row: f32 -> f16 conversion + fp32 squared norm.
__global__ __launch_bounds__(256)
void disp_prep(const float* __restrict__ Z, _Float16* __restrict__ Zh,
               float* __restrict__ SQ, int D) {
  const int row  = blockIdx.x * 8 + (threadIdx.x >> 5);
  const int lane = threadIdx.x & 31;
  const float4* zp = reinterpret_cast<const float4*>(Z + row * D);
  float s = 0.0f;
  for (int q = lane; q < D / 4; q += 32) {
    float4 v = zp[q];
    v4h h = { (_Float16)v.x, (_Float16)v.y, (_Float16)v.z, (_Float16)v.w };
    *reinterpret_cast<v4h*>(Zh + row * D + q * 4) = h;
    s += v.x * v.x + v.y * v.y + v.z * v.z + v.w * v.w;
  }
  #pragma unroll
  for (int off = 16; off > 0; off >>= 1) s += __shfl_xor(s, off, 32);
  if (lane == 0) SQ[row] = s;
}

// Async global->LDS copy, GVS addressing: base in SGPRs, 32-bit byte offset in VGPR.
__device__ __forceinline__ void async_b128(unsigned lds_byte_off, const void* sbase,
                                           int vbyte_off) {
  asm volatile("global_load_async_to_lds_b128 %0, %1, %2"
               :: "v"(lds_byte_off), "v"(vbyte_off), "s"(sbase) : "memory");
}

template <bool PRE>
__global__ __launch_bounds__(NTHREADS)
void disp_loss_wmma(const float* __restrict__ Z, const _Float16* __restrict__ Zh,
                    const float* __restrict__ SQ, float* __restrict__ out,
                    int N, int D, float invCnt) {
  const int bj = blockIdx.x, bi = blockIdx.y;
  if (bj < bi) return;                    // upper-triangular tiles only (block-uniform)
  const int i0 = bi * TILE, j0 = bj * TILE;

  __shared__ __attribute__((aligned(16))) _Float16 smem[2 * BUFH];  // [buf][A|B]
  __shared__ float sqI[TILE];
  __shared__ float sqJ[TILE];
  __shared__ float blockSum;

  const int t       = threadIdx.x;
  const int lane    = t & 31;
  const int wave    = t >> 5;
  const int lane16  = lane & 15;
  const int halfsel = lane >> 4;          // 0: lanes 0-15, 1: lanes 16-31
  const int wm      = wave >> 2;          // 0..1  -> 64 M-rows per wave
  const int wn      = wave & 3;           // 0..3  -> 32 N-cols per wave

  if (t == 0) blockSum = 0.0f;

  if constexpr (PRE) {
    if (t < TILE) sqI[t] = SQ[i0 + t];
    else          sqJ[t - TILE] = SQ[j0 + (t - TILE)];
  } else {
    const int r    = t & 127;
    const int side = t >> 7;
    const int row  = (side ? j0 : i0) + r;
    const float4* zp = reinterpret_cast<const float4*>(Z + row * D);
    float s = 0.0f;
    for (int q = 0; q < D / 4; ++q) {
      float4 v = zp[q];
      s += v.x * v.x + v.y * v.y + v.z * v.z + v.w * v.w;
    }
    if (side) sqJ[r] = s; else sqI[r] = s;
  }

  const unsigned smemBase = (unsigned)(uintptr_t)(&smem[0]);   // LDS byte offset

  // Issue one stage's 8 async b128 copies (4 A-slab + 4 B-slab per thread).
  auto issue_stage = [&](int s, int buf) {
    const int k0 = s * BK;
    const unsigned bufA = smemBase + (unsigned)buf * (BUFH * 2);
    const unsigned bufB = bufA + TILE * LDSTR * 2;
    #pragma unroll
    for (int q = 0; q < 4; ++q) {
      int f   = t + NTHREADS * q;          // 0..1023
      int row = f >> 3;                    // 0..127
      int kv  = (f & 7) << 3;              // 0..56 step 8 halves (16B)
      unsigned loff = (unsigned)(row * LDSTR + kv) * 2;
      async_b128(bufA + loff, Zh, ((i0 + row) * D + k0 + kv) * 2);
      async_b128(bufB + loff, Zh, ((j0 + row) * D + k0 + kv) * 2);
    }
  };

  v8f acc[4][2] = {};
  const int stages = D / BK;               // 8

  if constexpr (PRE) issue_stage(0, 0);

  for (int s = 0; s < stages; ++s) {
    const _Float16* As;
    const _Float16* Bs;
    if constexpr (PRE) {
      __syncthreads();                     // all waves done with buf[(s+1)&1] reads
      if (s + 1 < stages) {
        issue_stage(s + 1, (s + 1) & 1);
        asm volatile("s_wait_asynccnt 0x8" ::: "memory");  // stage-s copies complete
      } else {
        asm volatile("s_wait_asynccnt 0x0" ::: "memory");
      }
      __syncthreads();                     // everyone's stage-s LDS data visible
      As = smem + (s & 1) * BUFH;
      Bs = As + TILE * LDSTR;
    } else {
      const int k0 = s * BK;
      __syncthreads();
      #pragma unroll
      for (int q = 0; q < 8; ++q) {
        int f   = t + NTHREADS * q;        // 0..2047
        int row = f >> 4;                  // 0..127
        int kv  = (f & 15) << 2;           // 0..60 step 4
        float4 a = *reinterpret_cast<const float4*>(Z + (i0 + row) * D + k0 + kv);
        float4 b = *reinterpret_cast<const float4*>(Z + (j0 + row) * D + k0 + kv);
        v4h ah = { (_Float16)a.x, (_Float16)a.y, (_Float16)a.z, (_Float16)a.w };
        v4h bh = { (_Float16)b.x, (_Float16)b.y, (_Float16)b.z, (_Float16)b.w };
        *reinterpret_cast<v4h*>(&smem[row * LDSTR + kv]) = ah;
        *reinterpret_cast<v4h*>(&smem[TILE * LDSTR + row * LDSTR + kv]) = bh;
      }
      __syncthreads();
      As = smem;
      Bs = smem + TILE * LDSTR;
    }

    #pragma unroll
    for (int c = 0; c < 2; ++c) {          // two 32-K WMMA steps per stage
      const int kc = c * 32;
      v16h afr[4], bfr[2];

      // A fragment: two 8-half runs/lane (lanes<16 at +0,+16; lanes>=16 at +8,+24);
      // presents K permuted {0-7,16-23 | 8-15,24-31}. B below uses the same
      // permutation, so the contraction is unchanged.
      #pragma unroll
      for (int mt = 0; mt < 4; ++mt) {
        int row  = wm * 64 + mt * 16 + lane16;
        int base = row * LDSTR + kc + 8 * halfsel;
        v8h lo = *reinterpret_cast<const v8h*>(&As[base]);
        v8h hi = *reinterpret_cast<const v8h*>(&As[base + 16]);
        afr[mt] = __builtin_shufflevector(lo, hi, 0,1,2,3,4,5,6,7,8,9,10,11,12,13,14,15);
      }
      #pragma unroll
      for (int nt = 0; nt < 2; ++nt) {
        int row = wn * 32 + nt * 16 + lane16;
        int blo = row * LDSTR + kc + (halfsel ? 8 : 0);
        int bhi = row * LDSTR + kc + (halfsel ? 24 : 16);
        v8h lo = *reinterpret_cast<const v8h*>(&Bs[blo]);
        v8h hi = *reinterpret_cast<const v8h*>(&Bs[bhi]);
        bfr[nt] = __builtin_shufflevector(lo, hi, 0,1,2,3,4,5,6,7,8,9,10,11,12,13,14,15);
      }

      #pragma unroll
      for (int mt = 0; mt < 4; ++mt)
        #pragma unroll
        for (int nt = 0; nt < 2; ++nt)
          acc[mt][nt] = __builtin_amdgcn_wmma_f32_16x16x32_f16(
              false, afr[mt], false, bfr[nt], (short)0, acc[mt][nt], false, false);
    }
  }

  // Epilogue: d2 = |zi|^2 + |zj|^2 - 2 g ; contrib = exp(-sqrt(max(d2,0)))
  // D layout: element r of lane -> (M = r + 8*halfsel, N = lane16) in a 16x16 tile.
  float lsum = 0.0f;
  const bool offdiag = (bi != bj);
  #pragma unroll
  for (int mt = 0; mt < 4; ++mt) {
    #pragma unroll
    for (int nt = 0; nt < 2; ++nt) {
      int   nl = wn * 32 + nt * 16 + lane16;
      float sj = sqJ[nl];
      #pragma unroll
      for (int r = 0; r < 8; ++r) {
        int   ml = wm * 64 + mt * 16 + 8 * halfsel + r;
        float g  = acc[mt][nt][r];
        float d2 = sqI[ml] + sj - 2.0f * g;
        float d  = __builtin_amdgcn_sqrtf(fmaxf(d2, 0.0f));
        float v  = __expf(-d);
        lsum += (offdiag || (ml < nl)) ? v : 0.0f;   // exclude diagonal & lower half
      }
    }
  }
  #pragma unroll
  for (int off = 16; off > 0; off >>= 1)
    lsum += __shfl_xor(lsum, off, 32);
  if (lane == 0) atomicAdd(&blockSum, lsum);
  __syncthreads();
  if (t == 0) atomicAdd(out, blockSum * invCnt);
}

extern "C" void kernel_launch(void* const* d_in, const int* in_sizes, int n_in,
                              void* d_out, int out_size, void* d_ws, size_t ws_size,
                              hipStream_t stream) {
  (void)n_in; (void)out_size;
  const float* Z = (const float*)d_in[0];
  float* out = (float*)d_out;

  const int D = 512;
  const int N = in_sizes[0] / D;                        // 8192
  const long long cnt = (long long)N * (N - 1) / 2;
  const float invCnt = (float)(1.0 / (double)cnt);
  const int nb = N / TILE;                              // 64

  disp_zero_out<<<dim3(1), dim3(1), 0, stream>>>(out);

  const size_t need = (size_t)N * D * sizeof(_Float16) + (size_t)N * sizeof(float);
  if (d_ws != nullptr && ws_size >= need) {
    _Float16* Zh = (_Float16*)d_ws;
    float*    SQ = (float*)((char*)d_ws + (size_t)N * D * sizeof(_Float16));
    disp_prep<<<dim3(N / 8), dim3(256), 0, stream>>>(Z, Zh, SQ, D);
    disp_loss_wmma<true><<<dim3(nb, nb), dim3(NTHREADS), 0, stream>>>(
        Z, Zh, SQ, out, N, D, invCnt);
  } else {
    disp_loss_wmma<false><<<dim3(nb, nb), dim3(NTHREADS), 0, stream>>>(
        Z, nullptr, nullptr, out, N, D, invCnt);
  }
}
